// WindowAttention_41240275976870
// MI455X (gfx1250) — matrix operand order repaired and checked by reading, not compiled, not run on hardware
//
#include <hip/hip_runtime.h>
#include <stdint.h>

// ---------------- constants ----------------
#define WSZ    8
#define HEADS  12
#define HD     32
#define DISP   4
#define BDIM   16
#define CDIM   384
#define INNER  384          // HEADS*HD
#define NQKV   1152         // 3*INNER
#define HW     64
#define TOKENS (BDIM*HW*HW) // 65536
#define NW     64           // windows per image
#define SEQ    64           // tokens per window
#define ATTN_SCALE 0.17677669529663689f

typedef __attribute__((ext_vector_type(16))) _Float16 v16h;
typedef __attribute__((ext_vector_type(8)))  float    v8f;

union Frag16 { v16h h; uint32_t u[8]; };

__device__ __forceinline__ uint16_t f2h_bits(float f) {
    _Float16 h = (_Float16)f;
    return __builtin_bit_cast(uint16_t, h);
}

// low 32 bits of a flat address into the LDS aperture == LDS byte offset
__device__ __forceinline__ uint32_t lds_off(const void* p) {
    return (uint32_t)(uintptr_t)p;
}

// async copy of 16 bytes global -> LDS (ASYNCcnt-tracked, no VGPR staging)
__device__ __forceinline__ void async_b128(uint32_t lds_addr, uint32_t goff,
                                           const void* sbase) {
    asm volatile("global_load_async_to_lds_b128 %0, %1, %2"
                 :: "v"(lds_addr), "v"(goff), "s"(sbase) : "memory");
}
__device__ __forceinline__ void wait_async0() {
    asm volatile("s_wait_asynccnt 0x0" ::: "memory");
}

// ---------------- kernel 0: weights fp32 -> f16 ----------------
__global__ void __launch_bounds__(256)
k_convert_weights(const float* __restrict__ wqkv, const float* __restrict__ wout,
                  uint16_t* __restrict__ wqkv_h, uint16_t* __restrict__ wout_h) {
    int i = blockIdx.x * 256 + threadIdx.x;
    const int nq = CDIM * NQKV;
    const int no = INNER * CDIM;
    if (i < nq) { wqkv_h[i] = f2h_bits(wqkv[i]); return; }
    i -= nq;
    if (i < no) wout_h[i] = f2h_bits(wout[i]);
}

// ---------------- kernel 1: cyclic shift + CHW->token-major + f16 ----------------
// xs[token, c] = f16( x[b, c, (y+4)&63, (x+4)&63] ), token = (b*64+y)*64+x
__global__ void __launch_bounds__(256)
k_build_tokens(const float* __restrict__ x, uint16_t* __restrict__ xs) {
    __shared__ float tile[32][33];
    const int by = blockIdx.x;          // b*64 + y
    const int b  = by >> 6, y = by & 63;
    const int c0 = blockIdx.y * 32;
    const int x0 = blockIdx.z * 32;
    const int sy = (y + DISP) & 63;
#pragma unroll
    for (int t = 0; t < 4; ++t) {
        int idx = threadIdx.x + t * 256;   // 0..1023
        int cl = idx >> 5, xl = idx & 31;
        int sx = (x0 + xl + DISP) & 63;
        tile[cl][xl] = x[(((size_t)b * CDIM + c0 + cl) << 12) + (sy << 6) + sx];
    }
    __syncthreads();
#pragma unroll
    for (int t = 0; t < 4; ++t) {
        int idx = threadIdx.x + t * 256;
        int xl = idx >> 5, cl = idx & 31;
        xs[(((size_t)by << 6) + x0 + xl) * CDIM + c0 + cl] = f2h_bits(tile[cl][xl]);
    }
}

// ---------------- kernel 2: QKV GEMM (65536x1152 = A(65536x384) * W(384x1152)) ----
// 256 threads = 8 waves; block tile 128(M) x 64(N); K double-buffered 32 at a time.
// A tiles stream in via global_load_async_to_lds_b128; B tiles prefetch through
// registers (they need an n-major->k-major transpose). q scaled by 1/sqrt(hd).
__global__ void __launch_bounds__(256)
k_qkv_gemm(const uint16_t* __restrict__ A, const uint16_t* __restrict__ W,
           uint16_t* __restrict__ q, uint16_t* __restrict__ k, uint16_t* __restrict__ v) {
    __shared__ uint16_t As[2][128][40];   // 80B rows: every 16B async chunk aligned
    __shared__ uint16_t Bs[2][64][34];    // transposed: Bs[buf][n][k]
    const int row0 = blockIdx.x * 128;
    const int col0 = blockIdx.y * 64;
    const int tid  = threadIdx.x;
    const int wave = tid >> 5, lane = tid & 31;
    const int hi   = lane >> 4, ln = lane & 15;
    const int KT   = CDIM / 32;

    uint16_t breg[8];

    auto issueA = [&](int kt, int buf) {   // 128x32 f16 tile = 512 x 16B chunks
#pragma unroll
        for (int i = 0; i < 2; ++i) {
            int c = tid + i * 256;
            int m = c >> 2, ko = (c & 3) * 8;
            async_b128(lds_off(&As[buf][m][ko]),
                       (uint32_t)(((row0 + m) * CDIM + kt * 32 + ko) * 2), A);
        }
    };
    auto loadB = [&](int kt) {             // 32x64 tile -> regs
#pragma unroll
        for (int t = 0; t < 8; ++t) {
            int idx = tid + t * 256;
            int kk = idx >> 6, n = idx & 63;
            breg[t] = W[(size_t)(kt * 32 + kk) * NQKV + col0 + n];
        }
    };
    auto storeB = [&](int buf) {           // regs -> Bs[buf][n][k]
#pragma unroll
        for (int t = 0; t < 8; ++t) {
            int idx = tid + t * 256;
            int kk = idx >> 6, n = idx & 63;
            Bs[buf][n][kk] = breg[t];
        }
    };

    v8f acc[4] = {v8f{}, v8f{}, v8f{}, v8f{}};

    issueA(0, 0);
    loadB(0);
    storeB(0);
    wait_async0();
    __syncthreads();

    for (int kt = 0; kt < KT; ++kt) {
        const int buf = kt & 1;
        if (kt + 1 < KT) { issueA(kt + 1, buf ^ 1); loadB(kt + 1); }
        Frag16 a;
#pragma unroll
        for (int p = 0; p < 8; ++p) {              // A 16x32 f16: contiguous k = 16*hi + s
            int kb = 2 * p + hi * 16;
            a.u[p] = *(const uint32_t*)&As[buf][wave * 16 + ln][kb];
        }
#pragma unroll
        for (int nt = 0; nt < 4; ++nt) {
            Frag16 bf;
#pragma unroll
            for (int p = 0; p < 8; ++p)            // B 32x16 f16: contiguous k = 16*hi + s, as A
                bf.u[p] = *(const uint32_t*)&Bs[buf][nt * 16 + ln][2 * p + hi * 16];
            acc[nt] = __builtin_amdgcn_wmma_f32_16x16x32_f16(
                false, a.h, false, bf.h, (short)0, acc[nt], false, false);
        }
        if (kt + 1 < KT) storeB(buf ^ 1);
        wait_async0();
        __syncthreads();
    }

#pragma unroll
    for (int nt = 0; nt < 4; ++nt) {
#pragma unroll
        for (int r = 0; r < 8; ++r) {
            int mloc  = wave * 16 + r + hi * 8;    // C/D layout: M = r + 8*hi
            int token = row0 + mloc;
            int col   = col0 + nt * 16 + ln;
            int part  = col / INNER;
            int cc    = col - part * INNER;
            int head  = cc >> 5, d = cc & 31;
            int b   = token >> 12;
            int rem = token & 4095;
            int yy  = rem >> 6, xx = rem & 63;
            int wi  = ((yy >> 3) << 3) + (xx >> 3);
            int si  = ((yy & 7) << 3) + (xx & 7);
            size_t dst = ((((size_t)b * HEADS + head) * NW + wi) * SEQ + si) * HD + d;
            float val = acc[nt][r];
            if      (part == 0) q[dst] = f2h_bits(val * ATTN_SCALE);
            else if (part == 1) k[dst] = f2h_bits(val);
            else                v[dst] = f2h_bits(val);
        }
    }
}

// ---------------- kernel 3: windowed attention ----------------
// one block (4 waves, 128 thr) per (b, head, window); wave handles 16 query rows.
__global__ void __launch_bounds__(128)
k_attention(const uint16_t* __restrict__ q, const uint16_t* __restrict__ k,
            const uint16_t* __restrict__ v, const float* __restrict__ pos_emb,
            uint16_t* __restrict__ o) {
    __shared__ uint16_t Qs[64][40];     // async-loaded (16B-aligned rows)
    __shared__ uint16_t Ks[64][40];     // async-loaded
    __shared__ uint16_t Vt[32][66];     // V transposed: Vt[d][j] (sync path)
    __shared__ uint16_t Ps[64][66];     // exp probs (f16)
    __shared__ float pos[225];
    __shared__ float rinv[64];

    const int bid  = blockIdx.x;
    const int wi   = bid & 63;
    const int head = (bid >> 6) % HEADS;
    const int b    = bid / (64 * HEADS);
    const size_t base = (((size_t)b * HEADS + head) * NW + wi) * (SEQ * HD);
    const int tid = threadIdx.x;

    // Q/K: 64x32 f16 = 256 x 16B chunks, async straight into LDS
#pragma unroll
    for (int i = 0; i < 2; ++i) {
        int c = tid + i * 128;
        int r = c >> 2, dof = (c & 3) * 8;
        uint32_t go = (uint32_t)((base + (size_t)r * HD + dof) * 2);
        async_b128(lds_off(&Qs[r][dof]), go, q);
        async_b128(lds_off(&Ks[r][dof]), go, k);
    }
    // V needs a transpose -> sync path
#pragma unroll
    for (int t = 0; t < 16; ++t) {
        int idx = tid + t * 128;         // 0..2047
        int i = idx >> 5, d = idx & 31;
        Vt[d][i] = v[base + idx];
    }
    if (tid < 225) pos[tid] = pos_emb[tid];
    wait_async0();
    __syncthreads();

    const int wave = tid >> 5, lane = tid & 31;
    const int hi = lane >> 4, ln = lane & 15;
    const bool mUD = (wi >> 3) == 7;     // bottom row of windows
    const bool mLR = (wi & 7) == 7;      // right column of windows

    // ---- S = Q * K^T  (scale folded into q) : 4 WMMAs ----
    Frag16 a;
#pragma unroll
    for (int p = 0; p < 8; ++p) {
        int kb = 2 * p + hi * 16;
        a.u[p] = *(const uint32_t*)&Qs[wave * 16 + ln][kb];
    }
    v8f sacc[4] = {v8f{}, v8f{}, v8f{}, v8f{}};
#pragma unroll
    for (int nt = 0; nt < 4; ++nt) {
        Frag16 bf;
#pragma unroll
        for (int p = 0; p < 8; ++p)
            bf.u[p] = *(const uint32_t*)&Ks[nt * 16 + ln][2 * p + hi * 16];
        sacc[nt] = __builtin_amdgcn_wmma_f32_16x16x32_f16(
            false, a.h, false, bf.h, (short)0, sacc[nt], false, false);
    }

    // ---- bias + shift masks + row softmax (16-lane shfl reductions) ----
#pragma unroll
    for (int r = 0; r < 8; ++r) {
        int m = wave * 16 + r + hi * 8;           // query row
        float vals[4];
#pragma unroll
        for (int nt = 0; nt < 4; ++nt) {
            int j = nt * 16 + ln;                 // key col
            float s = sacc[nt][r] +
                      pos[((m >> 3) - (j >> 3) + 7) * 15 + ((m & 7) - (j & 7) + 7)];
            if (mUD && (((m >> 5) & 1) ^ ((j >> 5) & 1))) s = -1e30f;
            if (mLR && (((m >> 2) & 1) ^ ((j >> 2) & 1))) s = -1e30f;
            vals[nt] = s;
        }
        float mx = fmaxf(fmaxf(vals[0], vals[1]), fmaxf(vals[2], vals[3]));
#pragma unroll
        for (int off = 1; off < 16; off <<= 1) mx = fmaxf(mx, __shfl_xor(mx, off, 16));
        float sum = 0.f;
#pragma unroll
        for (int nt = 0; nt < 4; ++nt) {
            float e = __expf(vals[nt] - mx);
            sum += e;
            Ps[m][nt * 16 + ln] = f2h_bits(e);
        }
#pragma unroll
        for (int off = 1; off < 16; off <<= 1) sum += __shfl_xor(sum, off, 16);
        if (ln == 0) rinv[m] = 1.0f / sum;
    }
    __syncthreads();

    // ---- O = P * V : 2 k-chunks x 2 n-tiles = 4 WMMAs ----
    v8f oacc[2] = {v8f{}, v8f{}};
#pragma unroll
    for (int kc = 0; kc < 2; ++kc) {
        Frag16 pa;
#pragma unroll
        for (int p = 0; p < 8; ++p) {
            int kb = 2 * p + hi * 16 + kc * 32;
            pa.u[p] = *(const uint32_t*)&Ps[wave * 16 + ln][kb];
        }
#pragma unroll
        for (int nt = 0; nt < 2; ++nt) {
            Frag16 vb;
#pragma unroll
            for (int p = 0; p < 8; ++p)
                vb.u[p] = *(const uint32_t*)&Vt[nt * 16 + ln][2 * p + hi * 16 + kc * 32];
            oacc[nt] = __builtin_amdgcn_wmma_f32_16x16x32_f16(
                false, pa.h, false, vb.h, (short)0, oacc[nt], false, false);
        }
    }
    // ---- store token-major (fold 1/rowsum) ----
    const int wy = wi >> 3, wx = wi & 7;
#pragma unroll
    for (int nt = 0; nt < 2; ++nt) {
#pragma unroll
        for (int r = 0; r < 8; ++r) {
            int m = wave * 16 + r + hi * 8;
            int d = nt * 16 + ln;
            int token = (b << 12) + ((wy * 8 + (m >> 3)) << 6) + wx * 8 + (m & 7);
            o[(size_t)token * INNER + head * HD + d] = f2h_bits(oacc[nt][r] * rinv[m]);
        }
    }
}

// ---------------- kernel 4: out projection + bias + inverse roll + to (b,c,h,w) --
__global__ void __launch_bounds__(256)
k_out_gemm(const uint16_t* __restrict__ A, const uint16_t* __restrict__ W,
           const float* __restrict__ bias, float* __restrict__ out) {
    __shared__ uint16_t As[2][128][40];
    __shared__ uint16_t Bs[2][64][34];
    const int row0 = blockIdx.x * 128;
    const int col0 = blockIdx.y * 64;
    const int tid  = threadIdx.x;
    const int wave = tid >> 5, lane = tid & 31;
    const int hi   = lane >> 4, ln = lane & 15;
    const int KT   = INNER / 32;

    uint16_t breg[8];

    auto issueA = [&](int kt, int buf) {
#pragma unroll
        for (int i = 0; i < 2; ++i) {
            int c = tid + i * 256;
            int m = c >> 2, ko = (c & 3) * 8;
            async_b128(lds_off(&As[buf][m][ko]),
                       (uint32_t)(((row0 + m) * INNER + kt * 32 + ko) * 2), A);
        }
    };
    auto loadB = [&](int kt) {
#pragma unroll
        for (int t = 0; t < 8; ++t) {
            int idx = tid + t * 256;
            int kk = idx >> 6, n = idx & 63;
            breg[t] = W[(size_t)(kt * 32 + kk) * CDIM + col0 + n];
        }
    };
    auto storeB = [&](int buf) {
#pragma unroll
        for (int t = 0; t < 8; ++t) {
            int idx = tid + t * 256;
            int kk = idx >> 6, n = idx & 63;
            Bs[buf][n][kk] = breg[t];
        }
    };

    v8f acc[4] = {v8f{}, v8f{}, v8f{}, v8f{}};

    issueA(0, 0);
    loadB(0);
    storeB(0);
    wait_async0();
    __syncthreads();

    for (int kt = 0; kt < KT; ++kt) {
        const int buf = kt & 1;
        if (kt + 1 < KT) { issueA(kt + 1, buf ^ 1); loadB(kt + 1); }
        Frag16 a;
#pragma unroll
        for (int p = 0; p < 8; ++p) {
            int kb = 2 * p + hi * 16;
            a.u[p] = *(const uint32_t*)&As[buf][wave * 16 + ln][kb];
        }
#pragma unroll
        for (int nt = 0; nt < 4; ++nt) {
            Frag16 bf;
#pragma unroll
            for (int p = 0; p < 8; ++p)
                bf.u[p] = *(const uint32_t*)&Bs[buf][nt * 16 + ln][2 * p + hi * 16];
            acc[nt] = __builtin_amdgcn_wmma_f32_16x16x32_f16(
                false, a.h, false, bf.h, (short)0, acc[nt], false, false);
        }
        if (kt + 1 < KT) storeB(buf ^ 1);
        wait_async0();
        __syncthreads();
    }

#pragma unroll
    for (int nt = 0; nt < 4; ++nt) {
#pragma unroll
        for (int r = 0; r < 8; ++r) {
            int mloc  = wave * 16 + r + hi * 8;
            int token = row0 + mloc;
            int col   = col0 + nt * 16 + ln;
            int b  = token >> 12;
            int sy = (token >> 6) & 63, sx = token & 63;
            int oy = (sy + DISP) & 63, ox = (sx + DISP) & 63;   // inverse roll
            out[(((size_t)b * CDIM + col) << 12) + (oy << 6) + ox] = acc[nt][r] + bias[col];
        }
    }
}

// ---------------- host launcher ----------------
extern "C" void kernel_launch(void* const* d_in, const int* in_sizes, int n_in,
                              void* d_out, int out_size, void* d_ws, size_t ws_size,
                              hipStream_t stream) {
    (void)in_sizes; (void)n_in; (void)out_size; (void)ws_size;
    const float* x       = (const float*)d_in[0];
    const float* w_qkv   = (const float*)d_in[1];
    const float* w_out   = (const float*)d_in[2];
    const float* b_out   = (const float*)d_in[3];
    const float* pos_emb = (const float*)d_in[4];

    char* ws = (char*)d_ws;
    size_t off = 0;
    auto alloc = [&](size_t bytes) -> void* {
        void* p = ws + off;
        off += (bytes + 255) & ~(size_t)255;
        return p;
    };
    uint16_t* wqkv_h = (uint16_t*)alloc((size_t)CDIM * NQKV * 2);
    uint16_t* wout_h = (uint16_t*)alloc((size_t)INNER * CDIM * 2);
    uint16_t* xs     = (uint16_t*)alloc((size_t)TOKENS * CDIM * 2);
    uint16_t* qb     = (uint16_t*)alloc((size_t)TOKENS * INNER * 2);
    uint16_t* kb     = (uint16_t*)alloc((size_t)TOKENS * INNER * 2);
    uint16_t* vb     = (uint16_t*)alloc((size_t)TOKENS * INNER * 2);
    uint16_t* ob     = (uint16_t*)alloc((size_t)TOKENS * INNER * 2);

    const int ncvt = CDIM * NQKV + INNER * CDIM;
    k_convert_weights<<<(ncvt + 255) / 256, 256, 0, stream>>>(w_qkv, w_out, wqkv_h, wout_h);
    k_build_tokens<<<dim3(BDIM * HW, CDIM / 32, HW / 32), 256, 0, stream>>>(x, xs);
    k_qkv_gemm<<<dim3(TOKENS / 128, NQKV / 64), 256, 0, stream>>>(xs, wqkv_h, qb, kb, vb);
    k_attention<<<BDIM * HEADS * NW, 128, 0, stream>>>(qb, kb, vb, pos_emb, ob);
    k_out_gemm<<<dim3(TOKENS / 128, CDIM / 64), 256, 0, stream>>>(ob, wout_h, b_out, (float*)d_out);
}
